// DecoderSelfAttention_48825188221602
// MI455X (gfx1250) — matrix operand order repaired and checked by reading, not compiled, not run on hardware
//
#include <hip/hip_runtime.h>

// ---------------- types ----------------
typedef __attribute__((ext_vector_type(16))) __bf16 v16bf;
typedef __attribute__((ext_vector_type(8)))  __bf16 bf16x8;
typedef __attribute__((ext_vector_type(8)))  float  v8f;

union V16 { v16bf v; bf16x8 h2[2]; __bf16 e[16]; };

__device__ __forceinline__ v8f wmma_bf16(v16bf a, v16bf b, v8f c) {
  // D = A(16x32 bf16) * B(32x16 bf16) + C(16x16 f32)
  return __builtin_amdgcn_wmma_f32_16x16x32_bf16(false, a, false, b,
                                                 (short)0, c, false, false);
}

#define CDIM   768
#define C3     2304
#define HEADS  12
#define DHEAD  64

// ---------------- kernel 1: weight prep (transpose + bf16) ----------------
__global__ void prep_weights_k(const float* __restrict__ qkv_w,
                               const float* __restrict__ proj_w,
                               __bf16* __restrict__ wq_t,   // [2304][768]
                               __bf16* __restrict__ wp_t)   // [768][768]
{
  int i = blockIdx.x * blockDim.x + threadIdx.x;
  const int QT = C3 * CDIM;
  const int PT = CDIM * CDIM;
  if (i < QT) {
    int n = i / CDIM, k = i % CDIM;
    wq_t[i] = (__bf16)qkv_w[(size_t)k * C3 + n];
  } else if (i < QT + PT) {
    int j = i - QT;
    int n = j / CDIM, k = j % CDIM;
    wp_t[j] = (__bf16)proj_w[(size_t)k * CDIM + n];
  }
}

// ---------------- kernel 2: QKV GEMM (fp32 x -> bf16 qkv) ----------------
// wave: 16(M) x 64(N) tile; block: 8 waves stacked in M -> 128 x 64
__global__ __launch_bounds__(256) void qkv_gemm_k(
    const float* __restrict__ x,      // [N][768] fp32
    const __bf16* __restrict__ wt,    // [2304][768] bf16 (transposed)
    const float* __restrict__ bias,   // [2304]
    __bf16* __restrict__ out,         // [N][2304] bf16
    int Ntok)
{
  const int wid = threadIdx.x >> 5, lane = threadIdx.x & 31;
  const int lr = lane & 15, hi = lane >> 4;
  const int m0 = blockIdx.x * 128 + wid * 16;
  const int n0 = blockIdx.y * 64;

  int mrow = m0 + lr; if (mrow >= Ntok) mrow = Ntok - 1;
  const float* arow = x + (size_t)mrow * CDIM;

  v8f acc[4] = {};
  for (int k0 = 0; k0 < CDIM; k0 += 32) {
    V16 a;
    const float* p0 = arow + k0 + hi * 16;
#pragma unroll
    for (int j = 0; j < 8; ++j) a.e[j]     = (__bf16)p0[j];
#pragma unroll
    for (int j = 0; j < 8; ++j) a.e[8 + j] = (__bf16)p0[8 + j];
#pragma unroll
    for (int nn = 0; nn < 4; ++nn) {
      const __bf16* brow = wt + (size_t)(n0 + nn * 16 + lr) * CDIM + k0 + hi * 16;
      v16bf b = *(const v16bf*)brow;
      acc[nn] = wmma_bf16(a.v, b, acc[nn]);
    }
  }
#pragma unroll
  for (int nn = 0; nn < 4; ++nn) {
    const int n = n0 + nn * 16 + lr;
    const float bv = bias[n];
#pragma unroll
    for (int r = 0; r < 8; ++r) {
      int tok = m0 + r + hi * 8;
      if (tok < Ntok)
        out[(size_t)tok * C3 + n] = (__bf16)(acc[nn][r] + bv);
    }
  }
}

// ---------------- kernel 3: flash attention (one wave per seq/head/16-row tile) ----
__global__ __launch_bounds__(128) void attn_k(
    const __bf16* __restrict__ qkvb,  // [N][2304]: q|k|v each [H][D]
    __bf16* __restrict__ ob,          // [N][768]
    const int* __restrict__ cu,       // [nseq+1]
    int nseq)
{
  __shared__ __align__(16) __bf16 pbuf[4][16][32];

  const int wid = threadIdx.x >> 5, lane = threadIdx.x & 31;
  const int lr = lane & 15, hi = lane >> 4;
  const int t = blockIdx.x * 4 + wid;

  // map task -> (head, sequence, m-tile)
  int total = 0;
  for (int b = 0; b < nseq; ++b) {
    int Lb = cu[b + 1] - cu[b];
    total += (Lb + 15) >> 4;
  }
  if (t >= total * HEADS) return;
  int h = t / total;
  int r = t % total;
  int tok0 = 0, L = 0, mt = 0;
  for (int b = 0; b < nseq; ++b) {
    int Lb = cu[b + 1] - cu[b];
    int tb = (Lb + 15) >> 4;
    if (r < tb) { tok0 = cu[b]; L = Lb; mt = r; break; }
    r -= tb;
  }

  const __bf16* qp = qkvb + (size_t)tok0 * C3 + h * DHEAD;
  const __bf16* kp = qp + CDIM;
  const __bf16* vp = qp + 2 * CDIM;
  const int m0 = mt * 16;

  // Q fragments (16x64 = two 16x32 A-frags)
  int mrow = m0 + lr; if (mrow >= L) mrow = L - 1;
  V16 aq0, aq1;
  {
    const __bf16* qr = qp + (size_t)mrow * C3;
    aq0.h2[0] = *(const bf16x8*)(qr + hi * 16);
    aq0.h2[1] = *(const bf16x8*)(qr + hi * 16 + 8);
    aq1.h2[0] = *(const bf16x8*)(qr + 32 + hi * 16);
    aq1.h2[1] = *(const bf16x8*)(qr + 32 + hi * 16 + 8);
  }

  v8f o[4] = {};
  float mi[8], li[8];
#pragma unroll
  for (int rr = 0; rr < 8; ++rr) { mi[rr] = -1e30f; li[rr] = 0.0f; }

  __bf16 (*pt)[32] = pbuf[wid];
  const float scale = 0.125f;  // 1/sqrt(64)

  for (int l0 = 0; l0 < L; l0 += 32) {
    // ---- S = Q * K^T for columns [l0, l0+32) ----
    int n0r = l0 + lr;      if (n0r >= L) n0r = L - 1;
    int n1r = l0 + 16 + lr; if (n1r >= L) n1r = L - 1;
    const __bf16* kr0 = kp + (size_t)n0r * C3;
    const __bf16* kr1 = kp + (size_t)n1r * C3;
    v16bf bk00 = *(const v16bf*)(kr0 + hi * 16);
    v16bf bk01 = *(const v16bf*)(kr0 + 32 + hi * 16);
    v16bf bk10 = *(const v16bf*)(kr1 + hi * 16);
    v16bf bk11 = *(const v16bf*)(kr1 + 32 + hi * 16);

    v8f s0 = {}, s1 = {};
    s0 = wmma_bf16(aq0.v, bk00, s0);
    s0 = wmma_bf16(aq1.v, bk01, s0);
    s1 = wmma_bf16(aq0.v, bk10, s1);
    s1 = wmma_bf16(aq1.v, bk11, s1);

    const bool vld0 = (l0 + lr) < L;
    const bool vld1 = (l0 + 16 + lr) < L;
#pragma unroll
    for (int rr = 0; rr < 8; ++rr) {
      s0[rr] = vld0 ? s0[rr] * scale : -1e30f;
      s1[rr] = vld1 ? s1[rr] * scale : -1e30f;
    }

    // ---- online softmax (row stats across 16-lane halves) ----
#pragma unroll
    for (int rr = 0; rr < 8; ++rr) {
      float mx = fmaxf(s0[rr], s1[rr]);
      mx = fmaxf(mx, __shfl_xor(mx, 1));
      mx = fmaxf(mx, __shfl_xor(mx, 2));
      mx = fmaxf(mx, __shfl_xor(mx, 4));
      mx = fmaxf(mx, __shfl_xor(mx, 8));
      float mnew  = fmaxf(mi[rr], mx);
      float alpha = __expf(mi[rr] - mnew);
      float p0 = __expf(s0[rr] - mnew);
      float p1 = __expf(s1[rr] - mnew);
      float ls = p0 + p1;
      ls += __shfl_xor(ls, 1);
      ls += __shfl_xor(ls, 2);
      ls += __shfl_xor(ls, 4);
      ls += __shfl_xor(ls, 8);
      li[rr] = li[rr] * alpha + ls;
      mi[rr] = mnew;
      o[0][rr] *= alpha; o[1][rr] *= alpha;
      o[2][rr] *= alpha; o[3][rr] *= alpha;
      // stage P tile (C-layout -> LDS row-major 16x32)
      pt[rr + hi * 8][lr]      = (__bf16)p0;
      pt[rr + hi * 8][16 + lr] = (__bf16)p1;
    }

    // ---- reload P as A-fragment (16x32) ----
    V16 ap;
    ap.h2[0] = *(const bf16x8*)(&pt[lr][hi * 16]);
    ap.h2[1] = *(const bf16x8*)(&pt[lr][hi * 16 + 8]);

    // ---- O += P * V ----
#pragma unroll
    for (int nn = 0; nn < 4; ++nn) {
      V16 bv;
#pragma unroll
      for (int e = 0; e < 16; ++e) {
        int krow = l0 + hi * 16 + e; if (krow >= L) krow = L - 1;
        bv.e[e] = vp[(size_t)krow * C3 + nn * 16 + lr];
      }
      o[nn] = wmma_bf16(ap.v, bv.v, o[nn]);
    }
  }

  // ---- normalize + store bf16 ----
#pragma unroll
  for (int rr = 0; rr < 8; ++rr) {
    int row = m0 + rr + hi * 8;
    if (row < L) {
      float inv = 1.0f / li[rr];
      __bf16* orow = ob + (size_t)(tok0 + row) * CDIM + h * DHEAD;
      orow[lr]      = (__bf16)(o[0][rr] * inv);
      orow[16 + lr] = (__bf16)(o[1][rr] * inv);
      orow[32 + lr] = (__bf16)(o[2][rr] * inv);
      orow[48 + lr] = (__bf16)(o[3][rr] * inv);
    }
  }
}

// ---------------- kernel 4: output projection (bf16 -> fp32 d_out) ----------------
__global__ __launch_bounds__(256) void proj_gemm_k(
    const __bf16* __restrict__ xin,   // [N][768] bf16
    const __bf16* __restrict__ wt,    // [768][768] bf16 (transposed)
    const float* __restrict__ bias,   // [768]
    float* __restrict__ out,          // [N][768] fp32
    int Ntok)
{
  const int wid = threadIdx.x >> 5, lane = threadIdx.x & 31;
  const int lr = lane & 15, hi = lane >> 4;
  const int m0 = blockIdx.x * 128 + wid * 16;
  const int n0 = blockIdx.y * 64;

  int mrow = m0 + lr; if (mrow >= Ntok) mrow = Ntok - 1;
  const __bf16* arow = xin + (size_t)mrow * CDIM;

  v8f acc[4] = {};
  for (int k0 = 0; k0 < CDIM; k0 += 32) {
    V16 a;
    const __bf16* p0 = arow + k0 + hi * 16;
    a.h2[0] = *(const bf16x8*)p0;
    a.h2[1] = *(const bf16x8*)(p0 + 8);
#pragma unroll
    for (int nn = 0; nn < 4; ++nn) {
      const __bf16* brow = wt + (size_t)(n0 + nn * 16 + lr) * CDIM + k0 + hi * 16;
      v16bf b = *(const v16bf*)brow;
      acc[nn] = wmma_bf16(a.v, b, acc[nn]);
    }
  }
#pragma unroll
  for (int nn = 0; nn < 4; ++nn) {
    const int n = n0 + nn * 16 + lr;
    const float bv = bias[n];
#pragma unroll
    for (int r = 0; r < 8; ++r) {
      int tok = m0 + r + hi * 8;
      if (tok < Ntok)
        out[(size_t)tok * CDIM + n] = acc[nn][r] + bv;
    }
  }
}

// ---------------- host launcher ----------------
extern "C" void kernel_launch(void* const* d_in, const int* in_sizes, int n_in,
                              void* d_out, int out_size, void* d_ws, size_t ws_size,
                              hipStream_t stream) {
  const float* x      = (const float*)d_in[0];
  const float* qkv_w  = (const float*)d_in[1];
  const float* qkv_b  = (const float*)d_in[2];
  const float* proj_w = (const float*)d_in[3];
  const float* proj_b = (const float*)d_in[4];
  const int*   cu     = (const int*)d_in[5];

  const int N    = in_sizes[0] / CDIM;     // packed tokens
  const int nseq = in_sizes[5] - 1;

  // workspace layout (bf16 buffers, 256B aligned)
  char* ws = (char*)d_ws;
  size_t off = 0;
  __bf16* wq_t = (__bf16*)(ws + off); off += (size_t)C3 * CDIM * 2;   off = (off + 255) & ~(size_t)255;
  __bf16* wp_t = (__bf16*)(ws + off); off += (size_t)CDIM * CDIM * 2; off = (off + 255) & ~(size_t)255;
  __bf16* qkvb = (__bf16*)(ws + off); off += (size_t)N * C3 * 2;      off = (off + 255) & ~(size_t)255;
  __bf16* obuf = (__bf16*)(ws + off);

  // 1) weight prep
  {
    int total = C3 * CDIM + CDIM * CDIM;
    prep_weights_k<<<(total + 255) / 256, 256, 0, stream>>>(qkv_w, proj_w, wq_t, wp_t);
  }
  // 2) QKV projection
  {
    dim3 grid((N + 127) / 128, C3 / 64);
    qkv_gemm_k<<<grid, 256, 0, stream>>>(x, wq_t, qkv_b, qkvb, N);
  }
  // 3) flash attention (upper bound on tile count; extra waves self-exit)
  {
    int ub_tiles = (N + 15) / 16 + nseq;
    int tasks = ub_tiles * HEADS;
    attn_k<<<(tasks + 3) / 4, 128, 0, stream>>>(qkvb, obuf, cu, nseq);
  }
  // 4) output projection
  {
    dim3 grid((N + 127) / 128, CDIM / 64);
    proj_gemm_k<<<grid, 256, 0, stream>>>(obuf, wp_t, proj_b, (float*)d_out, N);
  }
  (void)n_in; (void)out_size; (void)ws_size;
}
